// HDCore_38809324486988
// MI455X (gfx1250) — matrix operand order repaired and checked by reading, not compiled, not run on hardware
//
#include <hip/hip_runtime.h>
#include <hip/hip_bf16.h>

// ---------------- problem constants (all tile-exact except D) ----------------
#define NN 16384     // rows (feat)        : 1024 tiles of 16
#define CC 128       // inner dim          : 32 steps of K=4 (f32 WMMA)
#define DD 10000     // hyperdim           : 625 tiles of 16
#define KK 3         // classes
#define DP 10016     // D padded to multiple of 32 for the bf16 (K=32) pass
#define GP 626       // DP/16 : sign-bit groups (u16) per hv row
#define NTILES 1024  // NN/16
#define DTILES 625   // DD/16

typedef __attribute__((ext_vector_type(2)))  float          v2f;
typedef __attribute__((ext_vector_type(8)))  float          v8f;
typedef __attribute__((ext_vector_type(16))) __bf16         v16bf;
typedef __attribute__((ext_vector_type(16))) unsigned short v16u16;

__device__ __forceinline__ unsigned short f32_to_bf16_rne(float f) {
    unsigned int x = __builtin_bit_cast(unsigned int, f);
    unsigned int r = x + 0x7FFFu + ((x >> 16) & 1u);
    return (unsigned short)(r >> 16);
}

// ---------------------------------------------------------------------------
// Kernel 1: hv = feat @ proj_w^T via V_WMMA_F32_16X16X4_F32 (exact f32),
// quantize to sign bits (ballot-packed, 1 bit/elem -> 20.5MB, L2 resident),
// and accumulate per-class sums (register class-accumulators per lane-column,
// folded with LDS float atomics, stored without global atomics).
// Grid: 626 blocks (625 compute strips of 16 D-columns + 1 zero-pad strip),
// 256 threads = 8 waves; each wave walks 128 N-tiles.
// ---------------------------------------------------------------------------
__global__ __launch_bounds__(256) void hd_pass1(
    const float* __restrict__ feat, const float* __restrict__ projw,
    const int* __restrict__ labels, unsigned short* __restrict__ hvb,
    float* __restrict__ per_class) {
  __shared__ float lds_acc[KK][16];
  const int tid = threadIdx.x;
  const int b   = blockIdx.x;

  if (b >= DTILES) {  // zero-pad column group 625 (cols 10000..10015)
    for (int n = tid; n < NN; n += 256) hvb[(size_t)n * GP + DTILES] = 0;
    return;
  }

  if (tid < KK * 16) lds_acc[tid / 16][tid % 16] = 0.0f;
  __syncthreads();

  const int lane = tid & 31;
  const int w    = tid >> 5;   // wave in block
  const int h    = lane >> 4;  // lane half (K split for A/B operands)
  const int col  = lane & 15;  // M for A-rows, N-column for B/C/D
  const int d0   = b * 16;

  // B panel: proj_w rows d0..d0+15, all 128 C. B[k][j] = projw[d0+j][c0+k].
  // 16x4 f32 operand layout: VGPR0 holds K=(2h), VGPR1 holds K=(2h+1).
  v2f bm[32];
  {
    const float* brow = projw + (size_t)(d0 + col) * CC + 2 * h;
#pragma unroll
    for (int kk = 0; kk < 32; ++kk) bm[kk] = *(const v2f*)(brow + 4 * kk);
  }

  float acc0 = 0.f, acc1 = 0.f, acc2 = 0.f;

  for (int t = w; t < NTILES; t += 8) {
    const int n0 = t * 16;
    const float* arow = feat + (size_t)(n0 + col) * CC + 2 * h;
    v8f c = {0.f, 0.f, 0.f, 0.f, 0.f, 0.f, 0.f, 0.f};
#pragma unroll
    for (int kk = 0; kk < 32; ++kk) {
      v2f a = *(const v2f*)(arow + 4 * kk);
      c = __builtin_amdgcn_wmma_f32_16x16x4_f32(false, a, false, bm[kk],
                                                (short)0, c, false, false);
    }
    // Quantize {>=0 -> +1}. C/D layout: lanes0-15 row v, lanes16-31 row v+8.
#pragma unroll
    for (int v = 0; v < 8; ++v) {
      const bool pos = (c[v] >= 0.0f);
      const unsigned int mask = __builtin_amdgcn_ballot_w32(pos);
      if (lane == 0) {
        hvb[(size_t)(n0 + v) * GP + b]     = (unsigned short)(mask & 0xFFFFu);
        hvb[(size_t)(n0 + v + 8) * GP + b] = (unsigned short)(mask >> 16);
      }
      const float s  = pos ? 1.0f : -1.0f;
      const int lab  = labels[n0 + v + 8 * h];
      if (lab == 0)      acc0 += s;
      else if (lab == 1) acc1 += s;
      else               acc2 += s;
    }
  }

  atomicAdd(&lds_acc[0][col], acc0);
  atomicAdd(&lds_acc[1][col], acc1);
  atomicAdd(&lds_acc[2][col], acc2);
  __syncthreads();
  if (tid < KK * 16)
    per_class[(size_t)(tid / 16) * DD + d0 + (tid % 16)] = lds_acc[tid / 16][tid % 16];
}

// ---------------------------------------------------------------------------
// Kernel 2: updated = cw + 0.5*per_class; row L2-normalize; emit bf16 protos
// padded to 16 rows x DP cols (zeros) as the pass-3 B matrix. 1 block.
// ---------------------------------------------------------------------------
__global__ __launch_bounds__(256) void hd_pass2(
    const float* __restrict__ cw, const float* __restrict__ pc,
    unsigned short* __restrict__ protos) {
  __shared__ float red[256];
  __shared__ float invn[KK];
  const int tid = threadIdx.x;
  for (int k = 0; k < KK; ++k) {
    float s = 0.f;
    for (int d = tid; d < DD; d += 256) {
      const float u = cw[k * DD + d] + 0.5f * pc[k * DD + d];
      s += u * u;
    }
    red[tid] = s;
    __syncthreads();
    for (int st = 128; st > 0; st >>= 1) {
      if (tid < st) red[tid] += red[tid + st];
      __syncthreads();
    }
    if (tid == 0) invn[k] = 1.0f / fmaxf(sqrtf(red[0]), 1e-12f);
    __syncthreads();
  }
  for (int r = 0; r < 16; ++r) {
    for (int d = tid; d < DP; d += 256) {
      unsigned short o = 0;
      if (r < KK && d < DD) {
        const float u = cw[r * DD + d] + 0.5f * pc[r * DD + d];
        o = f32_to_bf16_rne(u * invn[r]);
      }
      protos[(size_t)r * DP + d] = o;
    }
  }
}

// ---------------------------------------------------------------------------
// Kernel 3: logits = hv @ protos^T via V_WMMA_F32_16X16X32_BF16.
// A expanded in-register from packed sign bits to exact +/-1 bf16; hv bits are
// L2-resident so this pass never touches HBM for the big operand.
// 1024 waves (128 blocks x 8), one 16-row N-tile each, 313 K=32 steps.
// ---------------------------------------------------------------------------
__global__ __launch_bounds__(256) void hd_pass3(
    const unsigned short* __restrict__ hvb,
    const unsigned short* __restrict__ protos, float* __restrict__ out) {
  const int lane = threadIdx.x & 31;
  const int wave = (blockIdx.x * 256 + threadIdx.x) >> 5;  // 0..1023
  const int h    = lane >> 4;
  const int col  = lane & 15;
  const int n0   = wave * 16;

  const unsigned short* aptr = hvb + (size_t)(n0 + col) * GP;      // A row bits
  const unsigned short* bptr = protos + (size_t)col * DP + 16 * h; // B column
  const int base0 = 16 * h;  // A layout matches B: half0 -> K{0..15}, half1 -> K{16..31}

  v8f c = {0.f, 0.f, 0.f, 0.f, 0.f, 0.f, 0.f, 0.f};
  for (int g = 0; g < DP / 32; ++g) {
    const unsigned int bits = *(const unsigned int*)(aptr + 2 * g);  // K=0..31 of this row
    v16u16 au;
#pragma unroll
    for (int i = 0; i < 8; ++i)
      au[i] = ((bits >> (base0 + i)) & 1u) ? (unsigned short)0x3F80 : (unsigned short)0xBF80;
#pragma unroll
    for (int i = 8; i < 16; ++i)
      au[i] = ((bits >> (base0 + i)) & 1u) ? (unsigned short)0x3F80 : (unsigned short)0xBF80;
    const v16bf a  = __builtin_bit_cast(v16bf, au);
    const v16u16 bu = *(const v16u16*)(bptr + 32 * g);  // 32B aligned contiguous
    const v16bf bb = __builtin_bit_cast(v16bf, bu);
    c = __builtin_amdgcn_wmma_f32_16x16x32_bf16(false, a, false, bb,
                                                (short)0, c, false, false);
  }
  if (col < KK) {
#pragma unroll
    for (int v = 0; v < 8; ++v)
      out[(size_t)(n0 + v + 8 * h) * KK + col] = c[v];
  }
}

// ---------------------------------------------------------------------------
// Workspace layout (d_ws), all offsets 32B+ aligned:
//   [0)                hv sign bits : NN*GP*2      = 20,512,768 B
//   [20,512,768)       per_class    : 3*DD*4       =    120,000 B (pad 120,064)
//   [20,632,832)       protos bf16  : 16*DP*2      =    320,512 B
// Total ~20.95 MB.
// ---------------------------------------------------------------------------
extern "C" void kernel_launch(void* const* d_in, const int* in_sizes, int n_in,
                              void* d_out, int out_size, void* d_ws, size_t ws_size,
                              hipStream_t stream) {
  const float* feat   = (const float*)d_in[0];
  const float* projw  = (const float*)d_in[1];
  const float* cw     = (const float*)d_in[2];
  const int*   labels = (const int*)d_in[3];
  float*       out    = (float*)d_out;

  char* ws = (char*)d_ws;
  unsigned short* hvb       = (unsigned short*)ws;
  float*          per_class = (float*)(ws + 20512768);
  unsigned short* protos    = (unsigned short*)(ws + 20512768 + 120064);

  hd_pass1<<<DTILES + 1, 256, 0, stream>>>(feat, projw, labels, hvb, per_class);
  hd_pass2<<<1, 256, 0, stream>>>(cw, per_class, protos);
  hd_pass3<<<NTILES / 8, 256, 0, stream>>>(hvb, protos, out);
}
